// ContrastiveMambaEncoder_44676249813264
// MI455X (gfx1250) — compile-verified, matrix operand order checked by reading, not run on hardware
//
#include <hip/hip_runtime.h>
#include <hip/hip_bf16.h>
#include <math.h>

// ---------------- model dims ----------------
constexpr int B_ = 32, S_ = 512;
constexpr int L_ = 12, H_ = 12, DH_ = 64, DB_ = 768, DFF_ = 3072;
constexpr int DM_ = 512, DST_ = 128, P_ = 64, DI_ = 1024;
constexpr int NH_ = DI_ / P_;              // 16
constexpr int DCONV_ = 4;
constexpr int CONV_DIM_ = DI_ + 2 * DST_;  // 1280
constexpr int DIP_ = 2 * DI_ + 2 * DST_ + NH_; // 2320
constexpr int ED_ = 256;
constexpr int MTOK = B_ * S_;              // 16384

static inline int npad128(int n) { return (n + 127) & ~127; }

// ---------------- CDNA5 vector types ----------------
typedef __attribute__((ext_vector_type(16))) __bf16 v16bf;
typedef __attribute__((ext_vector_type(8)))  __bf16 v8bf;
typedef __attribute__((ext_vector_type(4)))  __bf16 v4bf;
typedef __attribute__((ext_vector_type(8)))  float  v8f;

__device__ __forceinline__ v16bf load_frag(const __bf16* p) {
  // two 16B LDS reads (ds_load_b128): k = kb..kb+7 and kb+16..kb+23
  v8bf lo = *(const v8bf*)p;
  v8bf hi = *(const v8bf*)(p + 16);
  v16bf r;
#pragma unroll
  for (int i = 0; i < 8; ++i) { r[i] = lo[i]; r[i + 8] = hi[i]; }
  return r;
}

__device__ __forceinline__ v4bf pack4(float4 f) {
  v4bf r;                                   // 2x v_cvt_pk_bf16_f32
  r[0] = (__bf16)f.x; r[1] = (__bf16)f.y; r[2] = (__bf16)f.z; r[3] = (__bf16)f.w;
  return r;
}

__device__ __forceinline__ float block_sum_256(float v) {
#pragma unroll
  for (int m = 16; m > 0; m >>= 1) v += __shfl_xor(v, m, 32);
  __shared__ float red[8];
  int w = threadIdx.x >> 5;
  if ((threadIdx.x & 31) == 0) red[w] = v;
  __syncthreads();
  float s = 0.f;
#pragma unroll
  for (int i = 0; i < 8; ++i) s += red[i];
  __syncthreads();
  return s;
}

__device__ __forceinline__ float gelu_tanh(float u) {
  return 0.5f * u * (1.f + tanhf(0.7978845608028654f * (u + 0.044715f * u * u * u)));
}
__device__ __forceinline__ float silu_(float x) { return x / (1.f + __expf(-x)); }

// ------- one-time weight transpose+convert: W[K][N] fp32 -> Wt[Npad][K] bf16 -
// Pad rows (n >= N) are zero so the GEMM never needs an N-tail path.
__global__ __launch_bounds__(256) void transpose_cvt_kernel(
    const float* __restrict__ W, __bf16* __restrict__ Wt, int K, int N) {
  __shared__ float T[64][65];
  const int k0 = blockIdx.x * 64, n0 = blockIdx.y * 64;
  const int tid = threadIdx.x;
#pragma unroll
  for (int i = 0; i < 4; ++i) {           // read 64k x 64n fp32, coalesced in n
    int f = tid + i * 256; int k = f >> 4, n4 = (f & 15) * 4;
    float4 v;
    if (n0 + n4 + 3 < N) {
      v = *(const float4*)&W[(size_t)(k0 + k) * N + n0 + n4];
    } else {
      const float* src = &W[(size_t)(k0 + k) * N + n0 + n4];
      v.x = (n0 + n4 + 0 < N) ? src[0] : 0.f;
      v.y = (n0 + n4 + 1 < N) ? src[1] : 0.f;
      v.z = (n0 + n4 + 2 < N) ? src[2] : 0.f;
      v.w = (n0 + n4 + 3 < N) ? src[3] : 0.f;
    }
    T[k][n4] = v.x; T[k][n4 + 1] = v.y; T[k][n4 + 2] = v.z; T[k][n4 + 3] = v.w;
  }
  __syncthreads();
#pragma unroll
  for (int i = 0; i < 2; ++i) {           // write transposed bf16, coalesced in k
    int f = tid + i * 256; int n = f >> 3, k8 = (f & 7) * 8;
    v8bf o;
#pragma unroll
    for (int j = 0; j < 8; ++j) o[j] = (__bf16)T[k8 + j][n];
    *(v8bf*)&Wt[(size_t)(n0 + n) * K + k0 + k8] = o;
  }
}

// ---------------- embedding + layernorm ----------------
__global__ __launch_bounds__(256) void embed_ln_kernel(
    const int* __restrict__ ids, const float* __restrict__ tok,
    const float* __restrict__ pos, const float* __restrict__ typ,
    const float* __restrict__ g, const float* __restrict__ bb,
    float* __restrict__ out) {
  int t = blockIdx.x;
  int si = t % S_;
  int id = ids[t];
  float v[3]; float s = 0.f;
#pragma unroll
  for (int i = 0; i < 3; ++i) {
    int c = threadIdx.x + i * 256;
    v[i] = tok[(size_t)id * DB_ + c] + pos[(size_t)si * DB_ + c] + typ[c];
    s += v[i];
  }
  float mean = block_sum_256(s) * (1.f / DB_);
  float s2 = 0.f;
#pragma unroll
  for (int i = 0; i < 3; ++i) { float d = v[i] - mean; s2 += d * d; }
  float rstd = rsqrtf(block_sum_256(s2) * (1.f / DB_) + 1e-12f);
#pragma unroll
  for (int i = 0; i < 3; ++i) {
    int c = threadIdx.x + i * 256;
    out[(size_t)t * DB_ + c] = (v[i] - mean) * rstd * g[c] + bb[c];
  }
}

// ---------------- residual + layernorm (in-place safe) ----------------
__global__ __launch_bounds__(256) void residual_ln_kernel(
    const float* __restrict__ x, const float* __restrict__ d,
    const float* __restrict__ g, const float* __restrict__ bb,
    float* __restrict__ out) {
  int t = blockIdx.x;
  float v[3]; float s = 0.f;
#pragma unroll
  for (int i = 0; i < 3; ++i) {
    int c = threadIdx.x + i * 256;
    v[i] = x[(size_t)t * DB_ + c] + d[(size_t)t * DB_ + c];
    s += v[i];
  }
  float mean = block_sum_256(s) * (1.f / DB_);
  float s2 = 0.f;
#pragma unroll
  for (int i = 0; i < 3; ++i) { float dd = v[i] - mean; s2 += dd * dd; }
  float rstd = rsqrtf(block_sum_256(s2) * (1.f / DB_) + 1e-12f);
#pragma unroll
  for (int i = 0; i < 3; ++i) {
    int c = threadIdx.x + i * 256;
    out[(size_t)t * DB_ + c] = (v[i] - mean) * rstd * g[c] + bb[c];
  }
}

// ---------------- bf16 WMMA GEMM:  C = act(A[M,K] @ W[K,N] + bias) ----------
// A: fp32 [M,K] (converted during staging).  Wt: bf16 [Npad][K] pre-transposed,
// pad rows zero => no N-tail path.  K % 64 == 0.  M via row clamp + store guard.
// 8 waves (wave32) in a 2x4 grid; block tile 64x128; each wave owns a 2x2 grid
// of 16x16 accumulators => 2 ds_load_b128 per WMMA.  K-step 64, double-buffered
// LDS (stride 72 => 144B rows, 16B-aligned runs).
__global__ __launch_bounds__(256) void gemm_bf16_kernel(
    const float* __restrict__ A, const __bf16* __restrict__ Wt,
    const float* __restrict__ bias, float* __restrict__ C,
    int M, int N, int K, int act) {
  __shared__ __bf16 As[2][64 * 72];    // 18.4 KB
  __shared__ __bf16 Bs[2][128 * 72];   // 36.9 KB ([n][k])
  const int n0 = blockIdx.x * 128, m0 = blockIdx.y * 64;
  const int tid = threadIdx.x, lane = tid & 31, w = tid >> 5;
  const int wm = w >> 2, wn = w & 3;               // 2 x 4 wave grid
  const int ml = lane & 15, kb = (lane >> 4) << 3;

  float4 aR[4]; v8bf bR[4];
  auto loadg = [&](int k0) {
#pragma unroll
    for (int i = 0; i < 4; ++i) {       // A: 64x64 fp32, 1024 float4s
      int f = tid + i * 256; int r = f >> 4, c4 = (f & 15) * 4;
      int row = m0 + r; row = (row < M) ? row : (M - 1);   // branchless clamp
      aR[i] = *(const float4*)&A[(size_t)row * K + k0 + c4];
    }
#pragma unroll
    for (int i = 0; i < 4; ++i) {       // Wt: 128n x 64k bf16, 1024 v8bf chunks
      int f = tid + i * 256; int n = f >> 3, k8 = (f & 7) * 8;
      bR[i] = *(const v8bf*)&Wt[(size_t)(n0 + n) * K + k0 + k8];
    }
  };
  auto stores = [&](int buf) {
#pragma unroll
    for (int i = 0; i < 4; ++i) {
      int f = tid + i * 256; int r = f >> 4, c4 = (f & 15) * 4;
      *(v4bf*)&As[buf][r * 72 + c4] = pack4(aR[i]);   // ds_store_b64
    }
#pragma unroll
    for (int i = 0; i < 4; ++i) {       // pure copy: ds_store_b128
      int f = tid + i * 256; int n = f >> 3, k8 = (f & 7) * 8;
      *(v8bf*)&Bs[buf][n * 72 + k8] = bR[i];
    }
  };

  v8f acc[2][2] = {};
  loadg(0);
  stores(0);
  __syncthreads();
  int buf = 0;
  for (int k0 = 0; k0 < K; k0 += 64) {
    const bool more = (k0 + 64 < K);
    if (more) {
      __builtin_prefetch(&A[(size_t)(m0 + (tid >> 2)) * K + k0 + 128], 0, 1);
      loadg(k0 + 64);                   // issue next-tile loads early
    }
#pragma unroll
    for (int c = 0; c < 2; ++c) {       // compute on current buffer
      v16bf a0 = load_frag(&As[buf][(wm * 32 + ml) * 72 + c * 32 + kb]);
      v16bf a1 = load_frag(&As[buf][(wm * 32 + 16 + ml) * 72 + c * 32 + kb]);
      v16bf b0 = load_frag(&Bs[buf][(wn * 32 + ml) * 72 + c * 32 + kb]);
      v16bf b1 = load_frag(&Bs[buf][(wn * 32 + 16 + ml) * 72 + c * 32 + kb]);
      acc[0][0] = __builtin_amdgcn_wmma_f32_16x16x32_bf16(false, a0, false, b0,
                                                          (short)0, acc[0][0], false, false);
      acc[0][1] = __builtin_amdgcn_wmma_f32_16x16x32_bf16(false, a0, false, b1,
                                                          (short)0, acc[0][1], false, false);
      acc[1][0] = __builtin_amdgcn_wmma_f32_16x16x32_bf16(false, a1, false, b0,
                                                          (short)0, acc[1][0], false, false);
      acc[1][1] = __builtin_amdgcn_wmma_f32_16x16x32_bf16(false, a1, false, b1,
                                                          (short)0, acc[1][1], false, false);
    }
    if (more) stores(buf ^ 1);
    __syncthreads();
    buf ^= 1;
  }

#pragma unroll
  for (int sn = 0; sn < 2; ++sn) {
    int col = n0 + wn * 32 + sn * 16 + ml;
    if (col < N) {
      float bv = bias ? bias[col] : 0.f;
#pragma unroll
      for (int sm = 0; sm < 2; ++sm)
#pragma unroll
        for (int r = 0; r < 8; ++r) {
          int row = m0 + wm * 32 + sm * 16 + r + ((lane >> 4) << 3);
          if (row < M) {
            float v = acc[sm][sn][r] + bv;
            if (act == 1) v = gelu_tanh(v);
            C[(size_t)row * N + col] = v;
          }
        }
    }
  }
}

// ---------------- flash attention (per (b,h), 64-query tile, 4 waves) -------
__global__ __launch_bounds__(128) void attn_kernel(
    const float* __restrict__ Qp, const float* __restrict__ Kp,
    const float* __restrict__ Vp, const int* __restrict__ mask,
    float* __restrict__ Op) {
  __shared__ __bf16 Ks[64 * 72];      // [key][d]
  __shared__ __bf16 Vs[64 * 72];      // [d][key]  (transposed)
  __shared__ __bf16 Ps[4][16 * 72];   // per-wave P tile [m][key]
  __shared__ float  biasS[64];
  const int bh = blockIdx.y, b = bh / H_, h = bh % H_;
  const int q0 = blockIdx.x * 64;
  const int tid = threadIdx.x, lane = tid & 31, w = tid >> 5;
  const int ml = lane & 15, kb = (lane >> 4) << 3;

  // Q fragments (A-layout), pre-scaled by 1/sqrt(DH)=0.125
  v16bf qf[2];
  const float* qrow = Qp + (size_t)(b * S_ + q0 + w * 16 + ml) * DB_ + h * DH_;
#pragma unroll
  for (int c = 0; c < 2; ++c) {
    float4 a0 = *(const float4*)&qrow[c * 32 + kb];
    float4 a1 = *(const float4*)&qrow[c * 32 + kb + 4];
    float4 b0 = *(const float4*)&qrow[c * 32 + kb + 16];
    float4 b1 = *(const float4*)&qrow[c * 32 + kb + 20];
    v16bf t;
    t[0] = (__bf16)(a0.x * 0.125f); t[1] = (__bf16)(a0.y * 0.125f);
    t[2] = (__bf16)(a0.z * 0.125f); t[3] = (__bf16)(a0.w * 0.125f);
    t[4] = (__bf16)(a1.x * 0.125f); t[5] = (__bf16)(a1.y * 0.125f);
    t[6] = (__bf16)(a1.z * 0.125f); t[7] = (__bf16)(a1.w * 0.125f);
    t[8]  = (__bf16)(b0.x * 0.125f); t[9]  = (__bf16)(b0.y * 0.125f);
    t[10] = (__bf16)(b0.z * 0.125f); t[11] = (__bf16)(b0.w * 0.125f);
    t[12] = (__bf16)(b1.x * 0.125f); t[13] = (__bf16)(b1.y * 0.125f);
    t[14] = (__bf16)(b1.z * 0.125f); t[15] = (__bf16)(b1.w * 0.125f);
    qf[c] = t;
  }
  float mrow[8], lrow[8];
#pragma unroll
  for (int r = 0; r < 8; ++r) { mrow[r] = -1e30f; lrow[r] = 0.f; }
  v8f oacc[4] = {};

  for (int kt = 0; kt < S_; kt += 64) {
    __syncthreads();
#pragma unroll
    for (int i = 0; i < 8; ++i) {     // stage K and V^T: 1024 float4s each dir
      int f = tid + i * 128; int d4 = (f & 15) * 4, key = f >> 4;
      size_t row = (size_t)(b * S_ + kt + key) * DB_ + h * DH_ + d4;
      float4 kv = *(const float4*)&Kp[row];
      float4 vv = *(const float4*)&Vp[row];
      *(v4bf*)&Ks[key * 72 + d4] = pack4(kv);
      v4bf vp = pack4(vv);
      Vs[(d4 + 0) * 72 + key] = vp[0];
      Vs[(d4 + 1) * 72 + key] = vp[1];
      Vs[(d4 + 2) * 72 + key] = vp[2];
      Vs[(d4 + 3) * 72 + key] = vp[3];
    }
    if (tid < 64) biasS[tid] = (mask[b * S_ + kt + tid] > 0) ? 0.f : -1e9f;
    __syncthreads();

    float pv[4][8]; float tmax[8];
#pragma unroll
    for (int r = 0; r < 8; ++r) tmax[r] = -1e30f;
#pragma unroll
    for (int nt = 0; nt < 4; ++nt) {          // scores: Q @ K^T (chained K=64)
      v8f sa = {};
#pragma unroll
      for (int c = 0; c < 2; ++c) {
        v16bf bf = load_frag(&Ks[(nt * 16 + ml) * 72 + c * 32 + kb]);
        sa = __builtin_amdgcn_wmma_f32_16x16x32_bf16(false, qf[c], false, bf,
                                                     (short)0, sa, false, false);
      }
      float bsc = biasS[nt * 16 + ml];
#pragma unroll
      for (int r = 0; r < 8; ++r) {
        float s = sa[r] + bsc; pv[nt][r] = s; tmax[r] = fmaxf(tmax[r], s);
      }
    }
#pragma unroll
    for (int m = 8; m > 0; m >>= 1)           // row max across 16-lane halves
#pragma unroll
      for (int r = 0; r < 8; ++r) tmax[r] = fmaxf(tmax[r], __shfl_xor(tmax[r], m, 32));
    float alpha[8], psum[8];
#pragma unroll
    for (int r = 0; r < 8; ++r) {
      float nm = fmaxf(mrow[r], tmax[r]);
      alpha[r] = __expf(mrow[r] - nm); mrow[r] = nm; psum[r] = 0.f;
    }
#pragma unroll
    for (int nt = 0; nt < 4; ++nt)
#pragma unroll
      for (int r = 0; r < 8; ++r) {
        float p = __expf(pv[nt][r] - mrow[r]); psum[r] += p;
        Ps[w][(r + ((lane >> 4) << 3)) * 72 + nt * 16 + ml] = (__bf16)p;
      }
#pragma unroll
    for (int m = 8; m > 0; m >>= 1)
#pragma unroll
      for (int r = 0; r < 8; ++r) psum[r] += __shfl_xor(psum[r], m, 32);
#pragma unroll
    for (int r = 0; r < 8; ++r) lrow[r] = lrow[r] * alpha[r] + psum[r];
#pragma unroll
    for (int nt = 0; nt < 4; ++nt)
#pragma unroll
      for (int r = 0; r < 8; ++r) oacc[nt][r] *= alpha[r];
    // O += P @ V   (same-wave LDS store->load is in-order)
#pragma unroll
    for (int nt = 0; nt < 4; ++nt)
#pragma unroll
      for (int c = 0; c < 2; ++c) {
        v16bf af = load_frag(&Ps[w][ml * 72 + c * 32 + kb]);
        v16bf bf = load_frag(&Vs[(nt * 16 + ml) * 72 + c * 32 + kb]);
        oacc[nt] = __builtin_amdgcn_wmma_f32_16x16x32_bf16(false, af, false, bf,
                                                           (short)0, oacc[nt], false, false);
      }
  }
#pragma unroll
  for (int nt = 0; nt < 4; ++nt)
#pragma unroll
    for (int r = 0; r < 8; ++r) {
      size_t row = (size_t)(b * S_ + q0 + w * 16 + r + ((lane >> 4) << 3));
      Op[row * DB_ + h * DH_ + nt * 16 + ml] = oacc[nt][r] / lrow[r];
    }
}

// ---------------- depthwise causal conv (w=4) + bias + SiLU -----------------
__global__ __launch_bounds__(256) void conv_silu_kernel(
    const float* __restrict__ z, const float* __restrict__ cw,
    const float* __restrict__ cb, float* __restrict__ out) {
  int t = blockIdx.x, bi = t / S_, si = t % S_;
  int c = blockIdx.y * 256 + threadIdx.x;   // CONV_DIM = 1280 = 5*256
  float s = 0.f;
#pragma unroll
  for (int j = 0; j < DCONV_; ++j) {
    int sp = si - (DCONV_ - 1) + j;
    if (sp >= 0)
      s += z[(size_t)(bi * S_ + sp) * DIP_ + DI_ + c] * cw[c * DCONV_ + j];
  }
  out[(size_t)t * CONV_DIM_ + c] = silu_(s + cb[c]);
}

// ---------------- dt = softplus(raw + dt_bias), dA = exp(dt * -exp(A_log)) --
__global__ __launch_bounds__(256) void dt_kernel(
    const float* __restrict__ z, const float* __restrict__ dtbias,
    const float* __restrict__ Alog, float* __restrict__ dtb,
    float* __restrict__ dab) {
  int idx = blockIdx.x * 256 + threadIdx.x;
  if (idx < MTOK * NH_) {
    int tok = idx / NH_, h = idx % NH_;
    float raw = z[(size_t)tok * DIP_ + DI_ + CONV_DIM_ + h] + dtbias[h];
    float dt = (raw > 20.f) ? raw : log1pf(__expf(raw));
    dtb[idx] = dt;
    dab[idx] = __expf(dt * (-__expf(Alog[h])));
  }
}

// ---------------- selective-state scan: one block per (h, b) ----------------
__global__ __launch_bounds__(256) void scan_kernel(
    const float* __restrict__ xc, const float* __restrict__ dtb,
    const float* __restrict__ dab, const float* __restrict__ Dskip,
    float* __restrict__ y) {
  const int h = blockIdx.x, b = blockIdx.y;
  const int tid = threadIdx.x, p = tid >> 2, nq = tid & 3;
  float hst[32];
#pragma unroll
  for (int i = 0; i < 32; ++i) hst[i] = 0.f;
  __shared__ float Bs[DST_], Cs[DST_], xsS[P_], sc[2];
  const float dsk = Dskip[h];
  for (int t = 0; t < S_; ++t) {
    size_t tok = (size_t)b * S_ + t;
    __syncthreads();
    if (tid < DST_) {
      Bs[tid] = xc[tok * CONV_DIM_ + DI_ + tid];
      Cs[tid] = xc[tok * CONV_DIM_ + DI_ + DST_ + tid];
    } else if (tid < DST_ + P_) {
      xsS[tid - DST_] = xc[tok * CONV_DIM_ + h * P_ + (tid - DST_)];
    } else if (tid == DST_ + P_) {
      sc[0] = dtb[tok * NH_ + h]; sc[1] = dab[tok * NH_ + h];
    }
    __syncthreads();
    float dt = sc[0], dA = sc[1];
    float dtx = dt * xsS[p];
    float yv = 0.f;
#pragma unroll
    for (int i = 0; i < 32; ++i) {
      int n = nq * 32 + i;
      hst[i] = dA * hst[i] + dtx * Bs[n];
      yv += hst[i] * Cs[n];
    }
    yv += __shfl_xor(yv, 1, 32);
    yv += __shfl_xor(yv, 2, 32);
    if (nq == 0) y[(tok * NH_ + h) * P_ + p] = yv + dsk * xsS[p];
  }
}

// ---------------- gated RMSNorm:  y = RMSNorm(y * silu(z)) * g  (in place) --
__global__ __launch_bounds__(256) void gated_rms_kernel(
    const float* __restrict__ z, const float* __restrict__ g,
    float* __restrict__ y) {
  int t = blockIdx.x;
  float v[4]; float s2 = 0.f;
#pragma unroll
  for (int i = 0; i < 4; ++i) {
    int c = threadIdx.x + i * 256;
    float zz = z[(size_t)t * DIP_ + c];
    v[i] = y[(size_t)t * DI_ + c] * silu_(zz);
    s2 += v[i] * v[i];
  }
  float rinv = rsqrtf(block_sum_256(s2) * (1.f / DI_) + 1e-5f);
#pragma unroll
  for (int i = 0; i < 4; ++i) {
    int c = threadIdx.x + i * 256;
    y[(size_t)t * DI_ + c] = v[i] * rinv * g[c];
  }
}

// ---------------- mean pool over S ------------------------------------------
__global__ __launch_bounds__(256) void pool_kernel(const float* __restrict__ w,
                                                   float* __restrict__ out) {
  int b = blockIdx.y, d = blockIdx.x * 256 + threadIdx.x;  // DM=512 -> grid.x=2
  float s = 0.f;
  for (int t = 0; t < S_; ++t) s += w[((size_t)b * S_ + t) * DM_ + d];
  out[(size_t)b * DM_ + d] = s * (1.f / S_);
}

// ---------------- host side --------------------------------------------------
static void gemm(hipStream_t st, const float* A, const __bf16* Wt,
                 const float* bias, float* C, int M, int N, int K, int act) {
  dim3 g(npad128(N) / 128, (M + 63) / 64);
  gemm_bf16_kernel<<<g, 256, 0, st>>>(A, Wt, bias, C, M, N, K, act);
}
static void cvtw(hipStream_t st, const float* W, __bf16* Wt, int K, int N) {
  dim3 g(K / 64, npad128(N) / 64);
  transpose_cvt_kernel<<<g, 256, 0, st>>>(W, Wt, K, N);
}

extern "C" void kernel_launch(void* const* d_in, const int* in_sizes, int n_in,
                              void* d_out, int out_size, void* d_ws, size_t ws_size,
                              hipStream_t stream) {
  const int*   ids    = (const int*)d_in[0];
  const int*   amask  = (const int*)d_in[1];
  const float* tok    = (const float*)d_in[2];
  const float* pos    = (const float*)d_in[3];
  const float* typ    = (const float*)d_in[4];
  const float* elng   = (const float*)d_in[5];
  const float* elnb   = (const float*)d_in[6];
  const float* Wq     = (const float*)d_in[7];
  const float* bq     = (const float*)d_in[8];
  const float* Wk     = (const float*)d_in[9];
  const float* bk     = (const float*)d_in[10];
  const float* Wv     = (const float*)d_in[11];
  const float* bv     = (const float*)d_in[12];
  const float* Wo     = (const float*)d_in[13];
  const float* bo     = (const float*)d_in[14];
  const float* ln1g   = (const float*)d_in[15];
  const float* ln1b   = (const float*)d_in[16];
  const float* Wf1    = (const float*)d_in[17];
  const float* bf1    = (const float*)d_in[18];
  const float* Wf2    = (const float*)d_in[19];
  const float* bf2    = (const float*)d_in[20];
  const float* ln2g   = (const float*)d_in[21];
  const float* ln2b   = (const float*)d_in[22];
  const float* W_pb   = (const float*)d_in[23];
  const float* b_pb   = (const float*)d_in[24];
  const float* W_in   = (const float*)d_in[25];
  const float* conv_w = (const float*)d_in[26];
  const float* conv_b = (const float*)d_in[27];
  const float* dtbias = (const float*)d_in[28];
  const float* Alog   = (const float*)d_in[29];
  const float* Dsk    = (const float*)d_in[30];
  const float* normg  = (const float*)d_in[31];
  const float* W_out  = (const float*)d_in[32];
  const float* W_proj = (const float*)d_in[33];
  const float* b_proj = (const float*)d_in[34];
  float* out = (float*)d_out;

  // workspace layout (bytes, 256-aligned)
  char* base = (char*)d_ws;
  size_t off = 0;
  auto allocb = [&](size_t bytes) -> void* {
    void* p = (void*)(base + off);
    off = (off + bytes + 255) & ~(size_t)255;
    return p;
  };
  auto allocf = [&](size_t elems) -> float* { return (float*)allocb(elems * 4); };
  auto allocw = [&](size_t elems) -> __bf16* { return (__bf16*)allocb(elems * 2); };

  float* xa   = allocf((size_t)MTOK * DB_);        // activations (in-place LN)
  float* qb   = allocf((size_t)MTOK * DB_);        // Q, then attention output
  float* kbuf = allocf((size_t)MTOK * DB_);
  float* vbuf = allocf((size_t)MTOK * DB_);
  float* t1   = allocf((size_t)MTOK * DB_);        // residual delta
  float* big  = allocf((size_t)MTOK * DFF_);       // FFN mid; later zxbcdt (DIP<DFF)
  float* xm   = allocf((size_t)MTOK * DM_);        // xm; later W_out output
  float* xc   = allocf((size_t)MTOK * CONV_DIM_);  // conv output
  float* dtb  = allocf((size_t)MTOK * NH_);
  float* dab  = allocf((size_t)MTOK * NH_);
  float* yb   = allocf((size_t)MTOK * DI_);
  float* pooled = allocf((size_t)B_ * DM_);

  // bf16 pre-transposed weights: Wt[Npad][K]
  const size_t wsz  = (size_t)DB_ * DB_;                    // 768x768
  const size_t f1sz = (size_t)npad128(DFF_) * DB_;          // 3072x768
  const size_t f2sz = (size_t)DB_ * DFF_;                   // 768x3072
  __bf16* wtq  = allocw(wsz * L_);
  __bf16* wtk  = allocw(wsz * L_);
  __bf16* wtv  = allocw(wsz * L_);
  __bf16* wto  = allocw(wsz * L_);
  __bf16* wtf1 = allocw(f1sz * L_);
  __bf16* wtf2 = allocw(f2sz * L_);
  __bf16* wtpb = allocw((size_t)npad128(DM_) * DB_);
  __bf16* wtin = allocw((size_t)npad128(DIP_) * DM_);
  __bf16* wtou = allocw((size_t)npad128(DM_) * DI_);
  __bf16* wtpj = allocw((size_t)npad128(ED_) * DM_);
  (void)ws_size; (void)n_in; (void)in_sizes; (void)out_size;

  // ---- one-time weight transpose+convert (fp32 [K][N] -> bf16 [Npad][K]) ----
  for (int l = 0; l < L_; ++l) {
    cvtw(stream, Wq  + (size_t)l * DB_ * DB_,  wtq  + wsz  * l, DB_,  DB_);
    cvtw(stream, Wk  + (size_t)l * DB_ * DB_,  wtk  + wsz  * l, DB_,  DB_);
    cvtw(stream, Wv  + (size_t)l * DB_ * DB_,  wtv  + wsz  * l, DB_,  DB_);
    cvtw(stream, Wo  + (size_t)l * DB_ * DB_,  wto  + wsz  * l, DB_,  DB_);
    cvtw(stream, Wf1 + (size_t)l * DB_ * DFF_, wtf1 + f1sz * l, DB_,  DFF_);
    cvtw(stream, Wf2 + (size_t)l * DFF_ * DB_, wtf2 + f2sz * l, DFF_, DB_);
  }
  cvtw(stream, W_pb,   wtpb, DB_, DM_);
  cvtw(stream, W_in,   wtin, DM_, DIP_);
  cvtw(stream, W_out,  wtou, DI_, DM_);
  cvtw(stream, W_proj, wtpj, DM_, ED_);

  // ---- embeddings ----
  embed_ln_kernel<<<MTOK, 256, 0, stream>>>(ids, tok, pos, typ, elng, elnb, xa);

  // ---- BERT stack ----
  for (int l = 0; l < L_; ++l) {
    gemm(stream, xa, wtq + wsz * l, bq + l * DB_, qb,   MTOK, DB_, DB_, 0);
    gemm(stream, xa, wtk + wsz * l, bk + l * DB_, kbuf, MTOK, DB_, DB_, 0);
    gemm(stream, xa, wtv + wsz * l, bv + l * DB_, vbuf, MTOK, DB_, DB_, 0);
    dim3 ag(S_ / 64, B_ * H_);
    attn_kernel<<<ag, 128, 0, stream>>>(qb, kbuf, vbuf, amask, qb);  // O aliases Q
    gemm(stream, qb, wto + wsz * l, bo + l * DB_, t1, MTOK, DB_, DB_, 0);
    residual_ln_kernel<<<MTOK, 256, 0, stream>>>(xa, t1, ln1g + l * DB_, ln1b + l * DB_, xa);
    gemm(stream, xa,  wtf1 + f1sz * l, bf1 + l * DFF_, big, MTOK, DFF_, DB_, 1); // GELU
    gemm(stream, big, wtf2 + f2sz * l, bf2 + l * DB_,  t1,  MTOK, DB_, DFF_, 0);
    residual_ln_kernel<<<MTOK, 256, 0, stream>>>(xa, t1, ln2g + l * DB_, ln2b + l * DB_, xa);
  }

  // ---- Mamba head ----
  float* zbuf = big;  // reuse FFN scratch for zxbcdt [MTOK, DIP]
  gemm(stream, xa, wtpb, b_pb,    xm,   MTOK, DM_,  DB_, 0);
  gemm(stream, xm, wtin, nullptr, zbuf, MTOK, DIP_, DM_, 0);
  conv_silu_kernel<<<dim3(MTOK, CONV_DIM_ / 256), 256, 0, stream>>>(zbuf, conv_w, conv_b, xc);
  dt_kernel<<<(MTOK * NH_ + 255) / 256, 256, 0, stream>>>(zbuf, dtbias, Alog, dtb, dab);
  scan_kernel<<<dim3(NH_, B_), 256, 0, stream>>>(xc, dtb, dab, Dsk, yb);
  gated_rms_kernel<<<MTOK, 256, 0, stream>>>(zbuf, normg, yb);
  gemm(stream, yb, wtou, nullptr, xm, MTOK, DM_, DI_, 0);   // xm reused as W_out output
  pool_kernel<<<dim3(DM_ / 256, B_), 256, 0, stream>>>(xm, pooled);
  gemm(stream, pooled, wtpj, b_proj, out, B_, ED_, DM_, 0);
}
